// ContrastiveLossManual_69123203662050
// MI455X (gfx1250) — matrix operand order repaired and checked by reading, not compiled, not run on hardware
//
#include <hip/hip_runtime.h>
#include <math.h>

typedef __attribute__((ext_vector_type(16))) _Float16 v16h;
typedef __attribute__((ext_vector_type(8)))  _Float16 v8h;
typedef __attribute__((ext_vector_type(4)))  _Float16 v4h;
typedef __attribute__((ext_vector_type(8)))  float    v8f;

constexpr int   kB   = 256;       // bags
constexpr int   kN   = 256;       // instances per bag
constexpr int   kD   = 768;       // embedding dim
constexpr int   kKC  = kD / 32;   // 24 K-chunks of 32
constexpr float kEps  = 1e-8f;
constexpr float kInvT = 2.0f;     // 1 / TEMPERATURE

constexpr int kAStride = 40;      // halves per A row in LDS (pad 32->40: bank-conflict-free frags)

#ifndef __has_builtin
#define __has_builtin(x) 0
#endif
#if __has_builtin(__builtin_amdgcn_global_load_async_to_lds_b128)
#define HAVE_ASYNC_LDS 1
#else
#define HAVE_ASYNC_LDS 0
#endif

#if HAVE_ASYNC_LDS
// Probe-discovered parameter type: pointer to 16-byte int vector.
typedef __attribute__((vector_size(4 * sizeof(int)))) int async_v4i;
static __device__ __forceinline__ void async_copy_b128(const void* g, void* l) {
  __builtin_amdgcn_global_load_async_to_lds_b128(
      (__attribute__((address_space(1))) async_v4i*)g,
      (__attribute__((address_space(3))) async_v4i*)l, 0, 0);
}
#endif

static __device__ __forceinline__ void wait_async_zero() {
#if HAVE_ASYNC_LDS
#if __has_builtin(__builtin_amdgcn_s_wait_asynccnt)
  __builtin_amdgcn_s_wait_asynccnt(0);
#else
  asm volatile("s_wait_asynccnt 0" ::: "memory");
#endif
#endif
}

// Workspace layout:
//   [0, 393216)        : bag matrix, f16, pre-swizzled into WMMA B-fragment order
//                        Bp[((c*16 + t)*32 + lane)*16 + j] ; c=K-chunk, t=N-tile, j=0..15
//   [393216, 393224)   : float acc[2] = { sum mask*(log_num-log_den), sum mask }

__global__ void zero_acc_kernel(float* acc) {
  if (threadIdx.x < 2) acc[threadIdx.x] = 0.0f;
}

// One block per bag row: L2-normalize, convert to f16, scatter into the
// per-lane WMMA B-fragment layout so the GEMM's LDS staging is a pure memcpy.
__global__ void bag_prep_kernel(const float* __restrict__ bag,
                                _Float16* __restrict__ Bp) {
  const int k   = blockIdx.x;   // bag index = logits column
  const int tid = threadIdx.x;  // 256 threads
  __shared__ float red[256];
  float s = 0.0f;
  for (int d = tid; d < kD; d += 256) {
    float v = bag[k * kD + d];
    s += v * v;
  }
  red[tid] = s;
  __syncthreads();
  for (int off = 128; off > 0; off >>= 1) {
    if (tid < off) red[tid] += red[tid + off];
    __syncthreads();
  }
  const float inv = 1.0f / fmaxf(sqrtf(red[0]), kEps);
  const int t  = k >> 4;        // N-tile
  const int nl = k & 15;        // lane-within-group = column-in-tile
  for (int d = tid; d < kD; d += 256) {
    const float v  = bag[k * kD + d] * inv;
    const int   c  = d >> 5;          // K-chunk
    const int   kl = d & 31;          // K within chunk
    const int lane = (((kl >> 3) & 1) << 4) | nl;  // hi half of lanes holds K=8..15,24..31
    const int j    = (kl & 7) | ((kl >> 4) << 3);
    Bp[(((size_t)c * 16 + t) * 32 + lane) * 16 + j] = (_Float16)v;
  }
}

// 512 blocks x 256 threads (8 waves). Block owns 128 logits rows x 256 cols.
// Software-pipelined, double-buffered LDS: chunk k+1's A tile streams into
// registers and its B chunk streams asynchronously into LDS while chunk k's
// 16 WMMAs execute. Instance tensor is read exactly once (f32->f16 cvt with
// on-the-fly row sum-of-squares); epilogue fuses diagonal-excluded logsumexp.
__global__ __launch_bounds__(256) void gemm_loss_kernel(
    const float* __restrict__ inst, const _Float16* __restrict__ Bp,
    const int* __restrict__ mask, float* __restrict__ acc) {
  __shared__ __attribute__((aligned(16))) _Float16 As[2][128 * kAStride]; // 20 KB
  __shared__ __attribute__((aligned(16))) _Float16 Bs[2][8192];           // 32 KB
  __shared__ float rowSq[128];

  const int  tid     = threadIdx.x;
  const int  wave    = tid >> 5;
  const int  lane    = tid & 31;
  const int  hi      = lane >> 4;            // 0: lanes 0-15, 1: lanes 16-31
  const long rowBase = (long)blockIdx.x * 128;

  v8f C[16];
  const v8f zf = {0.f, 0.f, 0.f, 0.f, 0.f, 0.f, 0.f, 0.f};
#pragma unroll
  for (int t = 0; t < 16; ++t) C[t] = zf;

  float4 av[4];
  float  sq[4] = {0.f, 0.f, 0.f, 0.f};

  // Global -> registers for A chunk kc, accumulating per-row sum(x^2).
  auto loadA = [&](int kc) {
#pragma unroll
    for (int i = 0; i < 4; ++i) {
      const int idx = tid + i * 256;   // 0..1023 float4 slots, 8 per row
      const int r   = idx >> 3;
      const int c4  = idx & 7;
      av[i] = *(const float4*)(inst + (rowBase + r) * kD + kc * 32 + c4 * 4);
      sq[i] += av[i].x * av[i].x + av[i].y * av[i].y +
               av[i].z * av[i].z + av[i].w * av[i].w;
    }
  };
  // Registers -> LDS (f32 -> f16) for buffer `buf`.
  auto storeA = [&](int buf) {
#pragma unroll
    for (int i = 0; i < 4; ++i) {
      const int idx = tid + i * 256;
      const int r   = idx >> 3;
      const int c4  = idx & 7;
      v4h h = {(_Float16)av[i].x, (_Float16)av[i].y,
               (_Float16)av[i].z, (_Float16)av[i].w};
      *(v4h*)&As[buf][r * kAStride + c4 * 4] = h;
    }
  };
  // Global -> LDS for swizzled B chunk kc into buffer `buf`.
  auto loadB = [&](int kc, int buf) {
#if HAVE_ASYNC_LDS
#pragma unroll
    for (int i = 0; i < 4; ++i) {
      const _Float16* src = Bp + (size_t)kc * 8192 + (size_t)(tid + i * 256) * 8;
      _Float16*       dst = &Bs[buf][(tid + i * 256) * 8];
      async_copy_b128(src, dst);
    }
#else
    const v8h* src = (const v8h*)(Bp + (size_t)kc * 8192);
    v8h*       dst = (v8h*)&Bs[buf][0];
#pragma unroll
    for (int i = 0; i < 4; ++i) dst[tid + i * 256] = src[tid + i * 256];
#endif
  };

  // Pipeline prologue: stage chunk 0.
  loadA(0);
  loadB(0, 0);
  storeA(0);
  wait_async_zero();
  __syncthreads();

  int cur = 0;
  for (int kc = 0; kc < kKC; ++kc) {
    const bool more = (kc + 1 < kKC);
    if (more) {
      loadA(kc + 1);          // global latency hidden under the WMMAs below
      loadB(kc + 1, cur ^ 1); // async, directly into the other LDS buffer
    }

    // A fragment (16x32 f16): lane row = lane&15; lane<16 holds K 0-7,16-23;
    // lane>=16 holds K 8-15,24-31 -> two 16B LDS reads, conflict-free stride.
    union { v16h v; v8h h[2]; } Af;
    const _Float16* abase =
        &As[cur][(wave * 16 + (lane & 15)) * kAStride + hi * 8];
    Af.h[0] = *(const v8h*)(abase);
    Af.h[1] = *(const v8h*)(abase + 16);

#pragma unroll
    for (int t = 0; t < 16; ++t) {
      union { v16h v; v8h h[2]; } Bf;
      const _Float16* bbase = &Bs[cur][(t * 32 + lane) * 16];
      Bf.h[0] = *(const v8h*)(bbase);
      Bf.h[1] = *(const v8h*)(bbase + 8);
      C[t] = __builtin_amdgcn_wmma_f32_16x16x32_f16(
          /*neg_a=*/false, Af.v, /*neg_b=*/false, Bf.v,
          /*c_mod=*/(short)0, C[t], /*reuse_a=*/false, /*reuse_b=*/false);
    }

    if (more) {
      storeA(cur ^ 1);        // cvt + ds_store after compute issue
      wait_async_zero();      // this wave's async B chunk is in LDS
    }
    __syncthreads();
    cur ^= 1;
  }

  // Finish per-row ||x||^2 in LDS.
  if (tid < 128) rowSq[tid] = 0.0f;
  __syncthreads();
#pragma unroll
  for (int i = 0; i < 4; ++i) atomicAdd(&rowSq[(tid + i * 256) >> 3], sq[i]);
  __syncthreads();

  // Epilogue: C layout (16x16 f32): VGPR r -> M = r + 8*hi ; col = t*16 + (lane&15).
  float lossAcc = 0.0f, maskAcc = 0.0f;
  for (int r = 0; r < 8; ++r) {
    const int  lr = wave * 16 + r + 8 * hi;     // local row in block
    const long R  = rowBase + lr;               // global logits row
    const float sc = kInvT / fmaxf(sqrtf(rowSq[lr]), kEps);
    const int  b     = (int)(R >> 8);           // diagonal column (bag index)
    const int  diagT = b >> 4, diagL = b & 15;
    const bool own   = (lane & 15) == diagL;

    float lmax = -INFINITY, lnum = 0.0f;
#pragma unroll
    for (int t = 0; t < 16; ++t) {
      const float v = C[t][r] * sc;
      if (own && t == diagT) lnum = v;          // numerator; excluded from lse
      else lmax = fmaxf(lmax, v);
    }
    for (int m = 1; m < 16; m <<= 1) lmax = fmaxf(lmax, __shfl_xor(lmax, m, 32));
    float se = 0.0f;
#pragma unroll
    for (int t = 0; t < 16; ++t) {
      if (own && t == diagT) continue;
      se += __expf(C[t][r] * sc - lmax);
    }
    for (int m = 1; m < 16; m <<= 1) se   += __shfl_xor(se, m, 32);
    for (int m = 1; m < 16; m <<= 1) lnum += __shfl_xor(lnum, m, 32);

    if ((lane & 15) == 0) {
      const float mk = (float)mask[R];
      lossAcc += mk * (lnum - (lmax + __logf(se)));
      maskAcc += mk;
    }
  }
  if ((lane & 15) == 0) {
    atomicAdd(&acc[0], lossAcc);
    atomicAdd(&acc[1], maskAcc);
  }
}

__global__ void finalize_kernel(const float* __restrict__ acc,
                                float* __restrict__ out) {
  out[0] = -acc[0] / acc[1];
}

extern "C" void kernel_launch(void* const* d_in, const int* in_sizes, int n_in,
                              void* d_out, int out_size, void* d_ws,
                              size_t ws_size, hipStream_t stream) {
  const float* inst = (const float*)d_in[0];  // [256,256,768] f32
  const float* bag  = (const float*)d_in[1];  // [256,768] f32
  const int*   mask = (const int*)d_in[2];    // [256,256] i32
  float* out = (float*)d_out;

  char*     ws  = (char*)d_ws;
  _Float16* Bp  = (_Float16*)ws;                 // 393216 bytes
  float*    acc = (float*)(ws + 393216);         // 8 bytes

  zero_acc_kernel<<<1, 32, 0, stream>>>(acc);
  bag_prep_kernel<<<kB, 256, 0, stream>>>(bag, Bp);
  gemm_loss_kernel<<<(kB * kN) / 128, 256, 0, stream>>>(inst, Bp, mask, acc);
  finalize_kernel<<<1, 1, 0, stream>>>(acc, out);
}
